// CrossAttention_40896678593196
// MI455X (gfx1250) — compile-verified, matrix operand order checked by reading, not run on hardware
//
#include <hip/hip_runtime.h>
#include <hip/hip_bf16.h>

#define BB   4
#define CC   256
#define NNQ  4096
#define BNR  (BB*NNQ)
#define DQK  32
#define QSCALE 0.17677669529663687f   // 1/sqrt(32)

typedef unsigned short u16t;
typedef __attribute__((ext_vector_type(16))) __bf16 v16bf;
typedef __attribute__((ext_vector_type(8)))  float  v8f;

union Frag16 { v16bf bf; uint4 q[2]; u16t h[16]; };
union U4     { uint4 v; u16t h[8]; };

__device__ __forceinline__ u16t f2bf(float f){
  unsigned u = __float_as_uint(f);
  unsigned r = (u + 0x7FFFu + ((u >> 16) & 1u)) >> 16;   // RNE fp32->bf16
  return (u16t)r;
}

// 16 bf16 halves at p[0..7] and p[16..23] (row-major A/B fragment: two K-half runs)
__device__ __forceinline__ v16bf frag_ld_split(const u16t* p){
  Frag16 f; f.q[0] = *(const uint4*)p; f.q[1] = *(const uint4*)(p + 16); return f.bf;
}
// 16 contiguous bf16 halves (pre-swizzled fragment layout)
__device__ __forceinline__ v16bf frag_ld_contig(const u16t* p){
  Frag16 f; f.q[0] = *(const uint4*)p; f.q[1] = *(const uint4*)(p + 8); return f.bf;
}
#define WMMA_BF16(a,b,c) __builtin_amdgcn_wmma_f32_16x16x32_bf16(false,(a),false,(b),(short)0,(c),false,false)

// Fragment-order linearization for a B operand tile (16 cols x 32 k):
//   slot(lane, j) holds element (k = 16*(j>>3) + 8*(lane>>4) + (j&7), col = lane&15)

// ---------------------------------------------------------------- converts
__global__ void k_cvt_inputs(const float* __restrict__ x, const float* __restrict__ y,
                             u16t* __restrict__ Xb, u16t* __restrict__ Yb){
  size_t tot = (size_t)BB*CC*NNQ;
  size_t i = (size_t)blockIdx.x*blockDim.x + threadIdx.x;
  if (i < tot)            Xb[i]       = f2bf(x[i]);
  else if (i < 2*tot)     Yb[i - tot] = f2bf(y[i - tot]);
}

// Wq/Wk/Wv row-major bf16; W1/W2 pre-swizzled into B-fragment order.
__global__ void k_cvt_weights(const float* __restrict__ Wq, const float* __restrict__ Wk,
                              const float* __restrict__ Wv, const float* __restrict__ W1,
                              const float* __restrict__ W2,
                              u16t* __restrict__ Wqb, u16t* __restrict__ Wkb,
                              u16t* __restrict__ Wvb, u16t* __restrict__ W1f,
                              u16t* __restrict__ W2f){
  int i = blockIdx.x*blockDim.x + threadIdx.x;
  if (i < 32*256){ Wqb[i] = f2bf(Wq[i]); Wkb[i] = f2bf(Wk[i]); }
  if (i < 256*256){
    Wvb[i] = f2bf(Wv[i]);
    int j = i & 15, lane = (i>>4)&31, kc = (i>>9)&7, ct = i>>12;
    int o = ct*16 + (lane & 15);
    int k = kc*32 + 16*(j>>3) + 8*(lane>>4) + (j&7);
    W1f[i] = f2bf(W1[o*256 + k]);
    W2f[i] = f2bf(W2[o*256 + k]);
  }
}

// ---------------------------------------------------------------- Q/K projection
// grid (NNQ/16, BB), block 128 (4 waves): wave 0,1 -> Q tiles, wave 2,3 -> K tiles.
// Q stored row-major [B][N][32] (pre-scaled, A-operand of QK^T).
// K stored in B-fragment order Kf[b][n>>4][lane][j].
__global__ void k_projqk(const u16t* __restrict__ Ain, const u16t* __restrict__ Bin,
                         const u16t* __restrict__ Wqb, const u16t* __restrict__ Wkb,
                         const float* __restrict__ bq, const float* __restrict__ bk,
                         u16t* __restrict__ Qb, u16t* __restrict__ Kf){
  __shared__ __attribute__((aligned(16))) u16t lta[16*32];   // [n][k] tile of q-source
  __shared__ __attribute__((aligned(16))) u16t ltb[16*32];   // [n][k] tile of k-source
  int tid = threadIdx.x;
  int lane = tid & 31, wid = tid >> 5, hk = (lane >> 4) & 1, l15 = lane & 15;
  int n0 = blockIdx.x * 16, b = blockIdx.y;

  const bool isQ = (wid < 2);
  const u16t* wrow = (isQ ? Wqb : Wkb) + (size_t)((wid & 1)*16 + l15)*256;
  const u16t* lt   = isQ ? lta : ltb;

  v8f acc = {0.f,0.f,0.f,0.f,0.f,0.f,0.f,0.f};
  for (int kc = 0; kc < 8; ++kc){
    __syncthreads();
    {   // stage transposed 32x16 chunks -> [n][k]; threads 0-63: a, 64-127: b
      int t2 = tid & 63;
      int kk = t2 >> 1, nb = (t2 & 1) * 8;
      size_t g = ((size_t)(b*CC + kc*32 + kk))*NNQ + n0 + nb;
      U4 u; u.v = *(const uint4*)((tid < 64 ? Ain : Bin) + g);
      u16t* dst = (tid < 64) ? lta : ltb;
      #pragma unroll
      for (int e = 0; e < 8; ++e) dst[(nb + e)*32 + kk] = u.h[e];
    }
    __syncthreads();
    v16bf wf = frag_ld_split(wrow + kc*32 + 8*hk);
    v16bf xf = frag_ld_split(lt + l15*32 + 8*hk);
    acc = WMMA_BF16(wf, xf, acc);
  }
  int n = n0 + l15;
  #pragma unroll
  for (int i = 0; i < 8; ++i){
    int d = (wid & 1)*16 + i + 8*hk;
    if (isQ){
      Qb[((size_t)(b*NNQ + n))*DQK + d] = f2bf((acc[i] + bq[d]) * QSCALE);
    } else {
      int lane2 = ((d >> 3) & 1)*16 + l15;
      int j2    = ((d >> 4) & 1)*8 + (d & 7);
      Kf[((size_t)(b*(NNQ/16) + (n0 >> 4))*32 + lane2)*16 + j2] = f2bf(acc[i] + bk[d]);
    }
  }
}

// ---------------------------------------------------------------- V projection
// grid (NNQ/16, BB), block 256 (8 waves); wave handles c-tiles wid*2, wid*2+1.
// V stored in PV B-fragment order Vf[b][m>>5][c>>4][lane][j] (k-dim = m).
__global__ void k_projv(const u16t* __restrict__ Bin, const u16t* __restrict__ Wvb,
                        const float* __restrict__ bv, u16t* __restrict__ Vf){
  __shared__ __attribute__((aligned(16))) u16t ltb[16*32];
  int tid = threadIdx.x;
  int lane = tid & 31, wid = tid >> 5, hk = (lane >> 4) & 1, l15 = lane & 15;
  int n0 = blockIdx.x * 16, b = blockIdx.y;

  const u16t* wrow0 = Wvb + (size_t)(wid*32      + l15)*256;
  const u16t* wrow1 = Wvb + (size_t)(wid*32 + 16 + l15)*256;

  v8f acc0 = {0.f,0.f,0.f,0.f,0.f,0.f,0.f,0.f};
  v8f acc1 = acc0;
  for (int kc = 0; kc < 8; ++kc){
    __syncthreads();
    if (tid < 64){
      int kk = tid >> 1, nb = (tid & 1) * 8;
      size_t g = ((size_t)(b*CC + kc*32 + kk))*NNQ + n0 + nb;
      U4 u; u.v = *(const uint4*)(Bin + g);
      #pragma unroll
      for (int e = 0; e < 8; ++e) ltb[(nb + e)*32 + kk] = u.h[e];
    }
    __syncthreads();
    v16bf xf = frag_ld_split(ltb + l15*32 + 8*hk);
    v16bf w0 = frag_ld_split(wrow0 + kc*32 + 8*hk);
    v16bf w1 = frag_ld_split(wrow1 + kc*32 + 8*hk);
    acc0 = WMMA_BF16(w0, xf, acc0);
    acc1 = WMMA_BF16(w1, xf, acc1);
  }
  // scatter into fragment order: m = n0+l15, 5 bits of m&31 -> (lane2 bit4, j2)
  int lane2m = ((l15 >> 3) & 1) * 16;
  int j2     = ((n0 >> 4) & 1) * 8 + (l15 & 7);
  size_t base = ((size_t)(b*(NNQ/32) + (n0 >> 5))) * 16;
  #pragma unroll
  for (int tt = 0; tt < 2; ++tt){
    v8f a = tt ? acc1 : acc0;
    #pragma unroll
    for (int i = 0; i < 8; ++i){
      int c = wid*32 + tt*16 + i + 8*hk;
      int lane2 = lane2m + (c & 15);
      Vf[((base + (c >> 4))*32 + lane2)*16 + j2] = f2bf(a[i] + bv[c]);
    }
  }
}

// ---------------------------------------------------------------- flash attention
// grid (NNQ/16, BB), block 128 (4 independent waves). Wave w owns channels [w*64, +64).
// All K/V fragments are direct 32B/lane global loads (L2-resident); only the
// P (D-layout -> A-layout) transpose goes through a per-wave LDS tile.
__global__ void k_attn(const u16t* __restrict__ Qb, const u16t* __restrict__ Kf,
                       const u16t* __restrict__ Vf, u16t* __restrict__ Ob){
  __shared__ __attribute__((aligned(16))) u16t ldsP[4*16*32];
  int tid = threadIdx.x;
  int lane = tid & 31, wid = tid >> 5, hk = (lane >> 4) & 1, l15 = lane & 15;
  int n0 = blockIdx.x * 16, b = blockIdx.y;

  v16bf qf = frag_ld_split(Qb + ((size_t)(b*NNQ + n0 + l15))*DQK + 8*hk);

  v8f zero = {0.f,0.f,0.f,0.f,0.f,0.f,0.f,0.f};
  v8f oacc[4] = {zero, zero, zero, zero};
  float mrow[8], lrow[8];
  #pragma unroll
  for (int i = 0; i < 8; ++i){ mrow[i] = -1e30f; lrow[i] = 0.f; }

  u16t* myP = ldsP + wid*(16*32);
  const u16t* kbase = Kf + ((size_t)(b*(NNQ/16))*32 + lane)*16;
  const u16t* vbase = Vf + (((size_t)(b*(NNQ/32))*16 + wid*4)*32 + lane)*16;

  for (int mc = 0; mc < NNQ/32; ++mc){
    v16bf kf0 = frag_ld_contig(kbase + (size_t)(2*mc    )*512);
    v16bf kf1 = frag_ld_contig(kbase + (size_t)(2*mc + 1)*512);
    v8f s0 = WMMA_BF16(qf, kf0, zero);
    v8f s1 = WMMA_BF16(qf, kf1, zero);
    if (mc + 1 < NNQ/32)
      __builtin_prefetch(vbase + (size_t)(mc + 1)*8192, 0, 1);

    float alpha[8];
    #pragma unroll
    for (int i = 0; i < 8; ++i){
      float mx = fmaxf(s0[i], s1[i]);
      #pragma unroll
      for (int off = 1; off < 16; off <<= 1) mx = fmaxf(mx, __shfl_xor(mx, off));
      float mnew = fmaxf(mrow[i], mx);
      float a  = __expf(mrow[i] - mnew);
      float p0 = __expf(s0[i] - mnew);
      float p1 = __expf(s1[i] - mnew);
      float s = p0 + p1;
      #pragma unroll
      for (int off = 1; off < 16; off <<= 1) s += __shfl_xor(s, off);
      mrow[i] = mnew; lrow[i] = lrow[i]*a + s; alpha[i] = a;
      int row = i + 8*hk;
      myP[row*32 + l15     ] = f2bf(p0);
      myP[row*32 + l15 + 16] = f2bf(p1);
    }
    asm volatile("" ::: "memory");   // LDS in-order within wave; fence compiler only
    v16bf pf = frag_ld_split(myP + l15*32 + 8*hk);
    #pragma unroll
    for (int ct = 0; ct < 4; ++ct){
      #pragma unroll
      for (int i = 0; i < 8; ++i) oacc[ct][i] *= alpha[i];
      v16bf vf = frag_ld_contig(vbase + (size_t)mc*8192 + ct*512);
      oacc[ct] = WMMA_BF16(pf, vf, oacc[ct]);
    }
  }
  #pragma unroll
  for (int ct = 0; ct < 4; ++ct){
    int c = wid*64 + ct*16 + l15;
    #pragma unroll
    for (int i = 0; i < 8; ++i){
      int n = n0 + i + 8*hk;
      Ob[((size_t)(b*NNQ + n))*CC + c] = f2bf(oacc[ct][i] / lrow[i]);
    }
  }
}

// ---------------------------------------------------------------- MLP GEMMs
// grid (BNR/16, 2), block 256 (8 waves); wave handles col tile ct = gy*8+wid.
__global__ void k_mlp1(const u16t* __restrict__ Ab, const u16t* __restrict__ Wf,
                       const float* __restrict__ bias, u16t* __restrict__ Hout){
  int tid = threadIdx.x;
  int lane = tid & 31, wid = tid >> 5, hk = (lane >> 4) & 1, l15 = lane & 15;
  int row0 = blockIdx.x * 16;
  int ct = blockIdx.y * 8 + wid;
  v8f acc = {0.f,0.f,0.f,0.f,0.f,0.f,0.f,0.f};
  const u16t* arow = Ab + (size_t)(row0 + l15)*CC;
  const u16t* wfrag = Wf + ((size_t)(ct*8)*32 + lane)*16;
  #pragma unroll
  for (int kc = 0; kc < 8; ++kc){
    v16bf af = frag_ld_split(arow + kc*32 + 8*hk);
    v16bf wf = frag_ld_contig(wfrag + (size_t)kc*512);
    acc = WMMA_BF16(af, wf, acc);
  }
  int o = ct*16 + l15;
  float bo = bias[o];
  #pragma unroll
  for (int i = 0; i < 8; ++i){
    int n = row0 + i + 8*hk;
    Hout[(size_t)n*CC + o] = f2bf(fmaxf(acc[i] + bo, 0.f));
  }
}

__global__ void k_mlp2(const u16t* __restrict__ Ab, const u16t* __restrict__ Wf,
                       const float* __restrict__ bias, float* __restrict__ Hout,
                       float* __restrict__ part){
  __shared__ float ws1[8], ws2[8];
  int tid = threadIdx.x;
  int lane = tid & 31, wid = tid >> 5, hk = (lane >> 4) & 1, l15 = lane & 15;
  int row0 = blockIdx.x * 16;
  int ct = blockIdx.y * 8 + wid;
  v8f acc = {0.f,0.f,0.f,0.f,0.f,0.f,0.f,0.f};
  const u16t* arow = Ab + (size_t)(row0 + l15)*CC;
  const u16t* wfrag = Wf + ((size_t)(ct*8)*32 + lane)*16;
  #pragma unroll
  for (int kc = 0; kc < 8; ++kc){
    v16bf af = frag_ld_split(arow + kc*32 + 8*hk);
    v16bf wf = frag_ld_contig(wfrag + (size_t)kc*512);
    acc = WMMA_BF16(af, wf, acc);
  }
  int o = ct*16 + l15;
  float bo = bias[o];
  float s1 = 0.f, s2 = 0.f;
  #pragma unroll
  for (int i = 0; i < 8; ++i){
    int n = row0 + i + 8*hk;
    float v = acc[i] + bo;
    Hout[(size_t)n*CC + o] = v;
    s1 += v; s2 += v*v;
  }
  #pragma unroll
  for (int off = 1; off < 32; off <<= 1){ s1 += __shfl_xor(s1, off); s2 += __shfl_xor(s2, off); }
  if (lane == 0){ ws1[wid] = s1; ws2[wid] = s2; }
  __syncthreads();
  if (tid == 0){
    float t1 = 0.f, t2 = 0.f;
    #pragma unroll
    for (int w = 0; w < 8; ++w){ t1 += ws1[w]; t2 += ws2[w]; }
    int pidx = blockIdx.x*2 + blockIdx.y;
    part[pidx*2 + 0] = t1;
    part[pidx*2 + 1] = t2;
  }
}

// deterministic per-batch reduction of 512 partials -> red[b][2]
__global__ void k_red(const float* __restrict__ part, float* __restrict__ red){
  __shared__ float sh1[8], sh2[8];
  int b = blockIdx.x, tid = threadIdx.x;
  int lane = tid & 31, wid = tid >> 5;
  float s1 = 0.f, s2 = 0.f;
  #pragma unroll
  for (int r = 0; r < 2; ++r){
    int e = b*512 + tid + 256*r;
    s1 += part[e*2 + 0]; s2 += part[e*2 + 1];
  }
  #pragma unroll
  for (int off = 1; off < 32; off <<= 1){ s1 += __shfl_xor(s1, off); s2 += __shfl_xor(s2, off); }
  if (lane == 0){ sh1[wid] = s1; sh2[wid] = s2; }
  __syncthreads();
  if (tid == 0){
    float t1 = 0.f, t2 = 0.f;
    #pragma unroll
    for (int w = 0; w < 8; ++w){ t1 += sh1[w]; t2 += sh2[w]; }
    red[b*2 + 0] = t1; red[b*2 + 1] = t2;
  }
}

// ---------------------------------------------------------------- LayerNorm + transpose out
__global__ void k_ln(const float* __restrict__ H2, const float* __restrict__ red,
                     const float* __restrict__ gamma, const float* __restrict__ beta,
                     float* __restrict__ out){
  size_t i = (size_t)blockIdx.x*blockDim.x + threadIdx.x;
  const size_t per = (size_t)CC*NNQ;
  if (i >= (size_t)BB*per) return;
  int b = (int)(i / per);
  size_t r = i % per;
  int c = (int)(r / NNQ), n = (int)(r % NNQ);
  float cnt  = (float)per;
  float mean = red[b*2] / cnt;
  float var  = red[b*2 + 1] / cnt - mean*mean;
  float rinv = rsqrtf(var + 1e-5f);
  float v = H2[((size_t)(b*NNQ + n))*CC + c];
  out[i] = (v - mean) * rinv * gamma[r] + beta[r];
}

// ---------------------------------------------------------------- host
extern "C" void kernel_launch(void* const* d_in, const int* in_sizes, int n_in,
                              void* d_out, int out_size, void* d_ws, size_t ws_size,
                              hipStream_t stream){
  (void)in_sizes; (void)n_in; (void)out_size; (void)ws_size;
  const float* x  = (const float*)d_in[0];
  const float* y  = (const float*)d_in[1];
  const float* Wq = (const float*)d_in[2];
  const float* bq = (const float*)d_in[3];
  const float* Wk = (const float*)d_in[4];
  const float* bk = (const float*)d_in[5];
  const float* Wv = (const float*)d_in[6];
  const float* bv = (const float*)d_in[7];
  const float* W1 = (const float*)d_in[8];
  const float* b1 = (const float*)d_in[9];
  const float* W2 = (const float*)d_in[10];
  const float* b2 = (const float*)d_in[11];
  const float* gamma = (const float*)d_in[12];
  const float* beta  = (const float*)d_in[13];
  float* out = (float*)d_out;

  char* ws = (char*)d_ws;
  size_t off = 0;
  auto alloc = [&](size_t bytes)->char*{
    char* p = ws + off; off += (bytes + 255) & ~(size_t)255; return p;
  };
  const size_t tot = (size_t)BB*CC*NNQ;
  u16t* Xb  = (u16t*)alloc(tot*2);
  u16t* Yb  = (u16t*)alloc(tot*2);
  u16t* Qb  = (u16t*)alloc((size_t)BB*NNQ*DQK*2);
  u16t* Kf  = (u16t*)alloc((size_t)BB*DQK*NNQ*2);   // fragment-order K
  u16t* Vf  = (u16t*)alloc(tot*2);                  // fragment-order V
  u16t* Obf = (u16t*)alloc(tot*2);
  u16t* H1  = (u16t*)alloc(tot*2);
  float* H2 = (float*)alloc(tot*4);
  u16t* Wqb = (u16t*)alloc(32*256*2);
  u16t* Wkb = (u16t*)alloc(32*256*2);
  u16t* Wvb = (u16t*)alloc(256*256*2);
  u16t* W1f = (u16t*)alloc(256*256*2);
  u16t* W2f = (u16t*)alloc(256*256*2);
  float* part = (float*)alloc(2048*2*4);
  float* red  = (float*)alloc(BB*2*4);

  k_cvt_inputs <<<dim3((unsigned)(2*tot/256)), 256, 0, stream>>>(x, y, Xb, Yb);
  k_cvt_weights<<<dim3(256), 256, 0, stream>>>(Wq, Wk, Wv, W1, W2, Wqb, Wkb, Wvb, W1f, W2f);

  for (int br = 0; br < 2; ++br){
    const u16t* A  = br ? Yb : Xb;   // q-source
    const u16t* Bv = br ? Xb : Yb;   // k/v-source
    k_projqk<<<dim3(NNQ/16, BB), 128, 0, stream>>>(A, Bv, Wqb, Wkb, bq, bk, Qb, Kf);
    k_projv <<<dim3(NNQ/16, BB), 256, 0, stream>>>(Bv, Wvb, bv, Vf);
    k_attn  <<<dim3(NNQ/16, BB), 128, 0, stream>>>(Qb, Kf, Vf, Obf);
    k_mlp1  <<<dim3(BNR/16, 2), 256, 0, stream>>>(Obf, W1f, b1, H1);
    k_mlp2  <<<dim3(BNR/16, 2), 256, 0, stream>>>(H1, W2f, b2, H2, part);
    k_red   <<<dim3(BB), 256, 0, stream>>>(part, red);
    k_ln    <<<dim3((unsigned)(tot/256)), 256, 0, stream>>>(H2, red, gamma, beta,
                                                            out + (size_t)br*tot);
  }
}
